// PointPillarScatter_Multi_pillar_od_34059090657828
// MI455X (gfx1250) — matrix operand order repaired and checked by reading, not compiled, not run on hardware
//
#include <hip/hip_runtime.h>
#include <hip/hip_bf16.h>

typedef __attribute__((ext_vector_type(16))) _Float16 v16h;
typedef __attribute__((ext_vector_type(8)))  float    v8f;

#define BATCH 2

// ---------------- utility kernels ----------------

__global__ void fill_h16_k(_Float16* p, long n) {
  long i = (long)blockIdx.x * blockDim.x + threadIdx.x;
  long s = (long)gridDim.x * blockDim.x;
  for (; i < n; i += s) p[i] = (_Float16)0.f;
}

__global__ void fill_f32_k(float* p, long n) {
  long i = (long)blockIdx.x * blockDim.x + threadIdx.x;
  long s = (long)gridDim.x * blockDim.x;
  for (; i < n; i += s) p[i] = 0.f;
}

// scatter point features [Np, C] f32 -> grid [B, C, H, W] f16
template <int C>
__global__ void scatter_h16_k(const float* __restrict__ f, const int* __restrict__ c,
                              _Float16* __restrict__ grid, long plane, int W, long total) {
  long i = (long)blockIdx.x * blockDim.x + threadIdx.x;
  if (i >= total) return;
  int ch = (int)(i % C);
  long pt = i / C;
  const int* cc = c + pt * 4;
  long pix = (long)cc[1] + (long)cc[2] * W + (long)cc[3];
  grid[((long)cc[0] * C + ch) * plane + pix] = (_Float16)f[pt * C + ch];
}

// scatter point features [Np, C] f32 -> f32 region (for add_features_to_map)
template <int C>
__global__ void scatter_f32_k(const float* __restrict__ f, const int* __restrict__ c,
                              float* __restrict__ grid, long plane, int W, long total) {
  long i = (long)blockIdx.x * blockDim.x + threadIdx.x;
  if (i >= total) return;
  int ch = (int)(i % C);
  long pt = i / C;
  const int* cc = c + pt * 4;
  long pix = (long)cc[1] + (long)cc[2] * W + (long)cc[3];
  grid[((long)cc[0] * C + ch) * plane + pix] = f[pt * C + ch];
}

// nearest resize fcyl [B,64,h,w] f32 -> d_out channels 64..127 of [B,128,H,W]
__global__ void resize_copy_k(const float* __restrict__ fc, float* __restrict__ outp,
                              int H, int W, int h, int w) {
  long total = (long)BATCH * 64 * H * W;
  long i = (long)blockIdx.x * blockDim.x + threadIdx.x;
  long s = (long)gridDim.x * blockDim.x;
  float ry = (float)h / (float)H, rx = (float)w / (float)W;
  for (; i < total; i += s) {
    int x = (int)(i % W); long r = i / W;
    int y = (int)(r % H); r /= H;
    int cch = (int)(r % 64); int b = (int)(r / 64);
    int ri = (int)((float)y * ry); if (ri >= h) ri = h - 1;
    int ci = (int)((float)x * rx); if (ci >= w) ci = w - 1;
    outp[(((long)b * 128 + 64 + cch) * H + y) * W + x] =
        fc[(((long)b * 64 + cch) * h + ri) * w + ci];
  }
}

// ---------------- weight packing (BN folded, WMMA A-fragment order) ----------------
// A operand as packed here (contiguous k, B uses the same): lane L holds row m=L&15, kbase=(L<16?0:16),
// element e -> k = kbase + e
// flat pack index = ((tile*nCh + kch)*32 + lane)*16 + e

__global__ void pack_conv_k(const float* __restrict__ w,
                            const float* __restrict__ bng, const float* __restrict__ bnv,
                            _Float16* __restrict__ wpk, int Cout, int Ktot) {
  int nCh = Ktot >> 5;
  long total = (long)(Cout / 16) * nCh * 512;
  long i = (long)blockIdx.x * blockDim.x + threadIdx.x;
  long st = (long)gridDim.x * blockDim.x;
  for (; i < total; i += st) {
    int e = (int)(i & 15), lane = (int)((i >> 4) & 31);
    long rest = i >> 9;
    int kch = (int)(rest % nCh);
    int tile = (int)(rest / nCh);
    int kb = (lane & 16) ? 16 : 0;
    int vv = e >> 1, p = e & 1;
    int k = (kch << 5) + kb + e;
    int co = (tile << 4) + (lane & 15);
    float sc = bng ? bng[co] * rsqrtf(bnv[co] + 1e-3f) : 1.f;
    wpk[i] = (_Float16)(w[(long)co * Ktot + k] * sc);
  }
}

__global__ void pack_bias_k(const float* g, const float* b, const float* m, const float* v,
                            const float* extra, float* biasOut, int Cout) {
  int co = blockIdx.x * blockDim.x + threadIdx.x;
  if (co < Cout) {
    float r = 0.f;
    if (g) { float s = g[co] * rsqrtf(v[co] + 1e-3f); r = b[co] - m[co] * s; }
    if (extra) r += extra[co];
    biasOut[co] = r;
  }
}

// deconv weight [64ci][64co][s][s] -> per-phase packs, Ktot=64, Cout=64
__global__ void pack_deconv_k(const float* __restrict__ w, _Float16* __restrict__ wpk, int s) {
  int phases = s * s;
  const long perPhase = 4 * 2 * 512;  // (64/16 tiles)*(64/32 kch)*512 = 4096
  long total = (long)phases * perPhase;
  long i = (long)blockIdx.x * blockDim.x + threadIdx.x;
  long st = (long)gridDim.x * blockDim.x;
  for (; i < total; i += st) {
    int phase = (int)(i / perPhase);
    long j = i % perPhase;
    int e = (int)(j & 15), lane = (int)((j >> 4) & 31);
    long rest = j >> 9;
    int kch = (int)(rest % 2);
    int tile = (int)(rest / 2);
    int kb = (lane & 16) ? 16 : 0;
    int vv = e >> 1, p = e & 1;
    int k = (kch << 5) + kb + e;  // = ci
    int co = (tile << 4) + (lane & 15);
    int py = phase / s, px = phase % s;
    wpk[i] = (_Float16)w[(((long)k * 64 + co) * s + py) * s + px];
  }
}

// ---------------- implicit-GEMM conv / deconv via WMMA ----------------
// block = 128 threads (4 waves). wave w computes out-channel tile [16w,16w+16)
// for ROWS rows x 16 output pixels. K = CIN*KSZ*KSZ (compile-time -> fully
// unrolled wmma chain; each A fragment is reused for ROWS wmmas).
// Deconv (kernel==stride==sS): phases>1, treat as 1x1 conv over input pixels,
// store at (ny*sS+py, nx*sS+px) with per-phase weight pack.

template <int KSZ, int CIN, int ROWS>
__global__ __launch_bounds__(128) void conv_wmma_k(
    const _Float16* __restrict__ src, int inH, int inW,
    const _Float16* __restrict__ wpk, long perPhaseW,
    const float* __restrict__ bias,
    const _Float16* __restrict__ res,
    _Float16* __restrict__ out16, float* __restrict__ out32,
    long outBS, long outCS,
    int outH, int outW, int nW, int nH,
    int stride, int pad,
    int relu, int sS, int phases) {
  constexpr int KK2 = KSZ * KSZ;
  constexpr int KTOT = CIN * KK2;
  constexpr int PANEL = 16 * KTOT;
  constexpr int NCH = KTOT / 32;
  extern __shared__ _Float16 lds[];

  const int z = blockIdx.z;
  const int phase = z % phases;
  const int b = z / phases;
  const int py = phase / sS, px = phase % sS;
  const int ny0 = blockIdx.y * ROWS;
  const int nxBase = blockIdx.x << 4;
  const int tid = threadIdx.x;

  const int wave = tid >> 5;
  const int lane = tid & 31;
  const _Float16* wp = wpk + (long)phase * perPhaseW;
  const _Float16* aBase = wp + (((long)wave * NCH) * 32 + lane) * 16;
  __builtin_prefetch(aBase, 0, 1);  // pull weight tile toward cache during staging

  // stage ROWS activation panels: lds[r*PANEL + n*KTOT + k]
  for (int idx = tid; idx < ROWS * PANEL; idx += 128) {
    int r = idx / PANEL;
    int j = idx - r * PANEL;
    int n = j & 15;
    int k = j >> 4;
    int ci = k / KK2;
    int rem = k - ci * KK2;
    int kh = rem / KSZ, kw = rem - kh * KSZ;
    int ny = ny0 + r;
    int iy = ny * stride - pad + kh;
    int ix = (nxBase + n) * stride - pad + kw;
    _Float16 vv = (_Float16)0.f;
    if (ny < nH && (nxBase + n) < nW && iy >= 0 && iy < inH && ix >= 0 && ix < inW)
      vv = src[(((long)b * CIN + ci) * inH + iy) * inW + ix];
    lds[r * PANEL + n * KTOT + k] = vv;
  }
  __syncthreads();

  const int nn = lane & 15;
  const int kbB = (lane & 16) ? 16 : 0;
  const _Float16* bRow = &lds[nn * KTOT + kbB];

  v8f acc[ROWS];
#pragma unroll
  for (int r = 0; r < ROWS; r++) acc[r] = (v8f){0.f, 0.f, 0.f, 0.f, 0.f, 0.f, 0.f, 0.f};

#pragma unroll
  for (int kc = 0; kc < NCH; kc++) {
    v16h a = *(const v16h*)(aBase + kc * 512);
#pragma unroll
    for (int r = 0; r < ROWS; r++) {
      v16h bm = *(const v16h*)(bRow + r * PANEL + kc * 32);
      acc[r] = __builtin_amdgcn_wmma_f32_16x16x32_f16(false, a, false, bm, (short)0,
                                                      acc[r], false, false);
    }
  }

  const int n = nxBase + nn;
  const int mBase = (wave << 4) + ((lane & 16) ? 8 : 0);
#pragma unroll
  for (int r = 0; r < ROWS; r++) {
    int ny = ny0 + r;
    if (ny < nH && n < nW) {
      int sy = ny * sS + py;
      int sx = n * sS + px;
#pragma unroll
      for (int q = 0; q < 8; q++) {
        int co = mBase + q;
        float a2 = acc[r][q];
        if (bias) a2 += bias[co];
        if (res)  a2 += (float)res[(((long)b * 64 + co) * outH + sy) * outW + sx];
        if (relu) a2 = fmaxf(a2, 0.f);
        long oi = (long)b * outBS + (long)co * outCS + (long)sy * outW + sx;
        if (out32) out32[oi] = a2;
        else       out16[oi] = (_Float16)a2;
      }
    }
  }
}

// ---------------- host orchestration ----------------

static inline size_t alignup(size_t x) { return (x + 255) & ~(size_t)255; }

template <int KSZ, int CIN, int ROWS>
static void launch_conv(hipStream_t s, const _Float16* src, int inH, int inW,
                        const _Float16* wpk, long perPhaseW, const float* bias,
                        const _Float16* res, _Float16* o16, float* o32,
                        long obs, long ocs, int outH, int outW, int nW, int nH,
                        int stride, int pad, int relu, int sS, int phases) {
  dim3 grid((nW + 15) / 16, (nH + ROWS - 1) / ROWS, BATCH * phases);
  size_t shmem = (size_t)ROWS * 16 * CIN * KSZ * KSZ * sizeof(_Float16);
  conv_wmma_k<KSZ, CIN, ROWS><<<grid, 128, shmem, s>>>(
      src, inH, inW, wpk, perPhaseW, bias, res, o16, o32, obs, ocs,
      outH, outW, nW, nH, stride, pad, relu, sS, phases);
}

// Per-SVN param layout after jax tree-flatten (sorted dict keys):
// 0:cb 1:cw 2:d2 3:d3
// res1(4): bn1{b,g,m,v}=4..7 bn2{b,g,m,v}=8..11 w1=12 w2=13
// res2(14): bn1=14..17 bn2=18..21 dbn=22..25 dw=26 w1=27 w2=28
// res3(29): bn1=29..32 bn2=33..36 dbn=37..40 dw=41 w1=42 w2=43
static void run_svn(hipStream_t st, void* const* P,
                    const _Float16* g, _Float16* cat, _Float16* t1, _Float16* t2,
                    _Float16* x2, _Float16* x3,
                    _Float16* wA, float* bA,
                    int H0, int W0, int H1, int W1, int H2, int W2,
                    float* outF32, long outBS, long outCS) {
  const long P0 = (long)H0 * W0, P1 = (long)H1 * W1, P2 = (long)H2 * W2;
  // carve weight arena (halves) and bias arena (floats)
  _Float16 *r1w1 = wA;           _Float16 *r1w2 = r1w1 + 36864;
  _Float16 *r2w1 = r1w2 + 36864; _Float16 *r2w2 = r2w1 + 36864;
  _Float16 *r2dw = r2w2 + 36864; _Float16 *r3w1 = r2dw + 4096;
  _Float16 *r3w2 = r3w1 + 36864; _Float16 *r3dw = r3w2 + 36864;
  _Float16 *d2p  = r3dw + 4096;  _Float16 *d3p  = d2p + 16384;
  _Float16 *cwp  = d3p + 65536;
  float *r1b1 = bA, *r1b2 = bA + 64, *r2b1 = bA + 128, *r2b2 = bA + 192, *r2bd = bA + 256;
  float *r3b1 = bA + 320, *r3b2 = bA + 384, *r3bd = bA + 448, *cbb = bA + 512;

  auto F = [](void* p) { return (const float*)p; };
  // pack weights + biases (BN folded)
  pack_conv_k<<<160, 256, 0, st>>>(F(P[12]), F(P[5]), F(P[7]), r1w1, 64, 576);
  pack_bias_k<<<1, 64, 0, st>>>(F(P[5]), F(P[4]), F(P[6]), F(P[7]), nullptr, r1b1, 64);
  pack_conv_k<<<160, 256, 0, st>>>(F(P[13]), F(P[9]), F(P[11]), r1w2, 64, 576);
  pack_bias_k<<<1, 64, 0, st>>>(F(P[9]), F(P[8]), F(P[10]), F(P[11]), nullptr, r1b2, 64);

  pack_conv_k<<<160, 256, 0, st>>>(F(P[27]), F(P[15]), F(P[17]), r2w1, 64, 576);
  pack_bias_k<<<1, 64, 0, st>>>(F(P[15]), F(P[14]), F(P[16]), F(P[17]), nullptr, r2b1, 64);
  pack_conv_k<<<160, 256, 0, st>>>(F(P[28]), F(P[19]), F(P[21]), r2w2, 64, 576);
  pack_bias_k<<<1, 64, 0, st>>>(F(P[19]), F(P[18]), F(P[20]), F(P[21]), nullptr, r2b2, 64);
  pack_conv_k<<<32, 256, 0, st>>>(F(P[26]), F(P[23]), F(P[25]), r2dw, 64, 64);
  pack_bias_k<<<1, 64, 0, st>>>(F(P[23]), F(P[22]), F(P[24]), F(P[25]), nullptr, r2bd, 64);

  pack_conv_k<<<160, 256, 0, st>>>(F(P[42]), F(P[30]), F(P[32]), r3w1, 64, 576);
  pack_bias_k<<<1, 64, 0, st>>>(F(P[30]), F(P[29]), F(P[31]), F(P[32]), nullptr, r3b1, 64);
  pack_conv_k<<<160, 256, 0, st>>>(F(P[43]), F(P[34]), F(P[36]), r3w2, 64, 576);
  pack_bias_k<<<1, 64, 0, st>>>(F(P[34]), F(P[33]), F(P[35]), F(P[36]), nullptr, r3b2, 64);
  pack_conv_k<<<32, 256, 0, st>>>(F(P[41]), F(P[38]), F(P[40]), r3dw, 64, 64);
  pack_bias_k<<<1, 64, 0, st>>>(F(P[38]), F(P[37]), F(P[39]), F(P[40]), nullptr, r3bd, 64);

  pack_deconv_k<<<64, 256, 0, st>>>(F(P[2]), d2p, 2);
  pack_deconv_k<<<256, 256, 0, st>>>(F(P[3]), d3p, 4);
  pack_conv_k<<<432, 256, 0, st>>>(F(P[1]), nullptr, nullptr, cwp, 64, 1728);
  pack_bias_k<<<1, 64, 0, st>>>(nullptr, nullptr, nullptr, nullptr, F(P[0]), cbb, 64);

  // res1: conv1 (g->t1, relu) ; conv2 (t1 + g -> cat[0:64], relu)
  launch_conv<3, 64, 4>(st, g, H0, W0, r1w1, 0, r1b1, nullptr, t1, nullptr,
                        64 * P0, P0, H0, W0, W0, H0, 1, 1, 1, 1, 1);
  launch_conv<3, 64, 4>(st, t1, H0, W0, r1w2, 0, r1b2, g, cat, nullptr,
                        192 * P0, P0, H0, W0, W0, H0, 1, 1, 1, 1, 1);
  // res2: dw (g->t2, stride2) ; conv1 (g->t1 s2, relu) ; conv2 (t1 + t2 -> x2, relu)
  launch_conv<1, 64, 4>(st, g, H0, W0, r2dw, 0, r2bd, nullptr, t2, nullptr,
                        64 * P1, P1, H1, W1, W1, H1, 2, 0, 0, 1, 1);
  launch_conv<3, 64, 4>(st, g, H0, W0, r2w1, 0, r2b1, nullptr, t1, nullptr,
                        64 * P1, P1, H1, W1, W1, H1, 2, 1, 1, 1, 1);
  launch_conv<3, 64, 4>(st, t1, H1, W1, r2w2, 0, r2b2, t2, x2, nullptr,
                        64 * P1, P1, H1, W1, W1, H1, 1, 1, 1, 1, 1);
  // res3: dw (x2->t2 s2) ; conv1 (x2->t1 s2, relu) ; conv2 (t1 + t2 -> x3, relu)
  launch_conv<1, 64, 4>(st, x2, H1, W1, r3dw, 0, r3bd, nullptr, t2, nullptr,
                        64 * P2, P2, H2, W2, W2, H2, 2, 0, 0, 1, 1);
  launch_conv<3, 64, 4>(st, x2, H1, W1, r3w1, 0, r3b1, nullptr, t1, nullptr,
                        64 * P2, P2, H2, W2, W2, H2, 2, 1, 1, 1, 1);
  launch_conv<3, 64, 4>(st, t1, H2, W2, r3w2, 0, r3b2, t2, x3, nullptr,
                        64 * P2, P2, H2, W2, W2, H2, 1, 1, 1, 1, 1);
  // deconvs into cat channels 64..127 and 128..191
  launch_conv<1, 64, 4>(st, x2, H1, W1, d2p, 4096, nullptr, nullptr, cat + 64 * P0, nullptr,
                        192 * P0, P0, H0, W0, W1, H1, 1, 0, 0, 2, 4);
  launch_conv<1, 64, 4>(st, x3, H2, W2, d3p, 4096, nullptr, nullptr, cat + 128 * P0, nullptr,
                        192 * P0, P0, H0, W0, W2, H2, 1, 0, 0, 4, 16);
  // final conv 192->64 (f32 out)
  launch_conv<3, 192, 2>(st, cat, H0, W0, cwp, 0, cbb, nullptr, nullptr, outF32,
                         outBS, outCS, H0, W0, W0, H0, 1, 1, 0, 1, 1);
}

extern "C" void kernel_launch(void* const* d_in, const int* in_sizes, int n_in,
                              void* d_out, int out_size, void* d_ws, size_t ws_size,
                              hipStream_t stream) {
  (void)in_sizes; (void)n_in; (void)out_size; (void)ws_size;
  // dims
  const int XH0 = 496, XW0 = 432, XH1 = 248, XW1 = 216, XH2 = 124, XW2 = 108;
  const int CH0 = 100, CW0 = 1280, CH1 = 50, CW1 = 640, CH2 = 25, CW2 = 320;
  const long XP0 = (long)XH0 * XW0, XP1 = (long)XH1 * XW1, XP2 = (long)XH2 * XW2;
  const long CP0 = (long)CH0 * CW0, CP1 = (long)CH1 * CW1, CP2 = (long)CH2 * CW2;
  const int NPXY = BATCH * 12000, NPCYL = BATCH * 8000;

  const float* xy_feat  = (const float*)d_in[0];
  const int*   xy_crd   = (const int*)d_in[1];
  const float* cyl_feat = (const float*)d_in[2];
  const int*   cyl_crd  = (const int*)d_in[3];
  const float* add_feat = (const float*)d_in[4];
  void* const* Pxy  = d_in + 5;
  void* const* Pcyl = d_in + 49;
  float* out = (float*)d_out;

  // workspace carve
  char* ws = (char*)d_ws;
  size_t off = 0;
  auto carve = [&](size_t bytes) { size_t r = off; off = alignup(off + bytes); return (void*)(ws + r); };
  _Float16* gxy  = (_Float16*)carve(BATCH * 64 * XP0 * 2);
  _Float16* catx = (_Float16*)carve(BATCH * 192 * XP0 * 2);
  _Float16* t1x  = (_Float16*)carve(BATCH * 64 * XP0 * 2);
  _Float16* t2x  = (_Float16*)carve(BATCH * 64 * XP1 * 2);
  _Float16* x2x  = (_Float16*)carve(BATCH * 64 * XP1 * 2);
  _Float16* x3x  = (_Float16*)carve(BATCH * 64 * XP2 * 2);
  _Float16* gcy  = (_Float16*)carve(BATCH * 64 * CP0 * 2);
  _Float16* catc = (_Float16*)carve(BATCH * 192 * CP0 * 2);
  _Float16* t1c  = (_Float16*)carve(BATCH * 64 * CP0 * 2);
  _Float16* t2c  = (_Float16*)carve(BATCH * 64 * CP1 * 2);
  _Float16* x2c  = (_Float16*)carve(BATCH * 64 * CP1 * 2);
  _Float16* x3c  = (_Float16*)carve(BATCH * 64 * CP2 * 2);
  float*    fcyl = (float*)carve(BATCH * 64 * CP0 * 4);
  _Float16* wAx  = (_Float16*)carve(450000 * 2);
  _Float16* wAc  = (_Float16*)carve(450000 * 2);
  float*    bAx  = (float*)carve(1024 * 4);
  float*    bAc  = (float*)carve(1024 * 4);

  // zero scatter targets
  {
    long n = BATCH * 64 * XP0;
    fill_h16_k<<<(int)((n + 255) / 256 > 65535 ? 65535 : (n + 255) / 256), 256, 0, stream>>>(gxy, n);
    n = BATCH * 64 * CP0;
    fill_h16_k<<<(int)((n + 255) / 256 > 65535 ? 65535 : (n + 255) / 256), 256, 0, stream>>>(gcy, n);
    long na = (long)BATCH * 3 * XP0;
    fill_f32_k<<<(int)((na + 255) / 256), 256, 0, stream>>>(out + (long)BATCH * 128 * XP0, na);
  }
  // scatter point features into BEV grids
  {
    long t = (long)NPXY * 64;
    scatter_h16_k<64><<<(int)((t + 255) / 256), 256, 0, stream>>>(xy_feat, xy_crd, gxy, XP0, XW0, t);
    t = (long)NPCYL * 64;
    scatter_h16_k<64><<<(int)((t + 255) / 256), 256, 0, stream>>>(cyl_feat, cyl_crd, gcy, CP0, CW0, t);
    t = (long)NPXY * 3;
    scatter_f32_k<3><<<(int)((t + 255) / 256), 256, 0, stream>>>(add_feat, xy_crd,
        out + (long)BATCH * 128 * XP0, XP0, XW0, t);
  }

  // SVN xy -> d_out channels [0,64)
  run_svn(stream, Pxy, gxy, catx, t1x, t2x, x2x, x3x, wAx, bAx,
          XH0, XW0, XH1, XW1, XH2, XW2, out, 128 * XP0, XP0);
  // SVN cyl -> fcyl
  run_svn(stream, Pcyl, gcy, catc, t1c, t2c, x2c, x3c, wAc, bAc,
          CH0, CW0, CH1, CW1, CH2, CW2, fcyl, 64 * CP0, CP0);
  // nearest resize fcyl into d_out channels [64,128)
  {
    long n = (long)BATCH * 64 * XP0;
    resize_copy_k<<<(int)((n + 255) / 256 > 65535 ? 65535 : (n + 255) / 256), 256, 0, stream>>>(
        fcyl, out, XH0, XW0, CH0, CW0);
  }
}
